// CDGP_44899588112462
// MI455X (gfx1250) — matrix operand order repaired and checked by reading, not compiled, not run on hardware
//
#include <hip/hip_runtime.h>
#include <hip/hip_bf16.h>

typedef __bf16 bf16_t;
typedef __attribute__((ext_vector_type(16))) __bf16 v16bf;
typedef __attribute__((ext_vector_type(8)))  __bf16 v8bf;
typedef __attribute__((ext_vector_type(8)))  float  v8f;

// Problem dimensions (fixed by reference)
constexpr int N_NODES = 200000;
constexpr int DM      = 500;
constexpr int B_EDGES = 4096;
constexpr int MSG     = 2 * DM + 1 + DM;   // 1501
constexpr int M_ROWS  = 2 * B_EDGES;       // 8192 message rows
constexpr int K_MSG   = 1536;              // MSG padded to multiple of 32
constexpr int K_H     = 512;               // DM padded to multiple of 32
constexpr int NP      = 1536;              // 3*DM=1500 padded to multiple of 64

// ---------------------------------------------------------------------------
// Fragment loader for one K-step (32 K-values).
//   A 16x32: lanes 0-15 row m hold K {0..15}; lanes 16-31 hold {16..31}
//   B 32x16: lane l holds 16 contiguous K of column (tn + l&15), K-offset 16*(l>>4)
// ---------------------------------------------------------------------------
template <int KP>
__device__ __forceinline__ void load_frags(const bf16_t* __restrict__ Arow,
                                           const bf16_t* __restrict__ Wrow,
                                           int kk, int half, v16bf& a, v16bf* b)
{
    a = *(const v16bf*)(Arow + kk + half * 16);
    const bf16_t* wp = Wrow + kk + half * 16;
    constexpr size_t WTILE = (size_t)16 * KP;   // W row stride between N tiles
#pragma unroll
    for (int j = 0; j < 4; ++j)
        b[j] = *(const v16bf*)(wp + (size_t)j * WTILE);
}

// ---------------------------------------------------------------------------
// Register-blocked, software-pipelined WMMA NT GEMM:
//   C[M, NP] = A[M, KP] * W[NP, KP]^T   (bf16 in, f32 accumulate/out)
// Per wave: 16(M) x 64(N), 4 accumulators; A fragment reused by 4 WMMAs.
// Double-buffered fragments: next K-step's 10 b128 loads are issued before the
// current step's WMMAs, so the wmma group waits only on the older loads
// (in-order LOADcnt) and overlaps with the in-flight ones.
// Per block: 4 waves stacked along M -> 64x64 block tile.
// C/D layout: VGPR v, lane l -> row v + 8*(l>>4), col l&15.
// ---------------------------------------------------------------------------
template <int KP>
__global__ __launch_bounds__(128) void wmma_gemm_nt(const bf16_t* __restrict__ A,
                                                    const bf16_t* __restrict__ W,
                                                    float* __restrict__ C)
{
    const int lane = threadIdx.x & 31;
    const int wave = threadIdx.x >> 5;
    const int half = lane >> 4;
    const int l16  = lane & 15;
    const int tm   = blockIdx.x * 64 + wave * 16;   // wave's M tile
    const int tn0  = blockIdx.y * 64;               // block's N range (4 tiles/wave)

    const bf16_t* __restrict__ Arow = A + (size_t)(tm + l16) * KP;
    const bf16_t* __restrict__ Wrow = W + (size_t)(tn0 + l16) * KP;

    v8f acc0 = {}, acc1 = {}, acc2 = {}, acc3 = {};

    v16bf a_cur, b_cur[4];
    load_frags<KP>(Arow, Wrow, 0, half, a_cur, b_cur);

#pragma unroll 2
    for (int kk = 32; kk < KP; kk += 32) {
        v16bf a_nxt, b_nxt[4];
        load_frags<KP>(Arow, Wrow, kk, half, a_nxt, b_nxt);

        acc0 = __builtin_amdgcn_wmma_f32_16x16x32_bf16(false, a_cur, false, b_cur[0], (short)0, acc0, false, false);
        acc1 = __builtin_amdgcn_wmma_f32_16x16x32_bf16(false, a_cur, false, b_cur[1], (short)0, acc1, false, false);
        acc2 = __builtin_amdgcn_wmma_f32_16x16x32_bf16(false, a_cur, false, b_cur[2], (short)0, acc2, false, false);
        acc3 = __builtin_amdgcn_wmma_f32_16x16x32_bf16(false, a_cur, false, b_cur[3], (short)0, acc3, false, false);

        a_cur = a_nxt;
#pragma unroll
        for (int j = 0; j < 4; ++j) b_cur[j] = b_nxt[j];
    }
    // epilogue: last K-step
    acc0 = __builtin_amdgcn_wmma_f32_16x16x32_bf16(false, a_cur, false, b_cur[0], (short)0, acc0, false, false);
    acc1 = __builtin_amdgcn_wmma_f32_16x16x32_bf16(false, a_cur, false, b_cur[1], (short)0, acc1, false, false);
    acc2 = __builtin_amdgcn_wmma_f32_16x16x32_bf16(false, a_cur, false, b_cur[2], (short)0, acc2, false, false);
    acc3 = __builtin_amdgcn_wmma_f32_16x16x32_bf16(false, a_cur, false, b_cur[3], (short)0, acc3, false, false);

    float* Crow = C + (size_t)(tm + half * 8) * NP + tn0 + l16;
#pragma unroll
    for (int v = 0; v < 8; ++v) {
        Crow[(size_t)v * NP]      = acc0[v];
        Crow[(size_t)v * NP + 16] = acc1[v];
        Crow[(size_t)v * NP + 32] = acc2[v];
        Crow[(size_t)v * NP + 48] = acc3[v];
    }
}

// ---------------------------------------------------------------------------
// Convert f32 weight [rows, cols] -> zero-padded bf16 [rows_p, cols_p]
// ---------------------------------------------------------------------------
__global__ void conv_w(const float* __restrict__ Wsrc, bf16_t* __restrict__ Wdst,
                       int rows, int cols, int rows_p, int cols_p)
{
    size_t total = (size_t)rows_p * cols_p;
    for (size_t idx = blockIdx.x * (size_t)blockDim.x + threadIdx.x; idx < total;
         idx += (size_t)gridDim.x * blockDim.x) {
        int r = (int)(idx / cols_p);
        int c = (int)(idx % cols_p);
        float v = (r < rows && c < cols) ? Wsrc[(size_t)r * cols + c] : 0.0f;
        Wdst[idx] = (bf16_t)v;
    }
}

// ---------------------------------------------------------------------------
// 'last' aggregation: per-node argmax over message position
// ---------------------------------------------------------------------------
__global__ void init_lastpos(int* __restrict__ lp)
{
    int i = blockIdx.x * blockDim.x + threadIdx.x;
    if (i < N_NODES) lp[i] = -1;
}

__global__ void scan_last(const int* __restrict__ src, const int* __restrict__ dst,
                          int* __restrict__ lp)
{
    int i = blockIdx.x * blockDim.x + threadIdx.x;
    if (i >= M_ROWS) return;
    int node = (i < B_EDGES) ? src[i] : dst[i - B_EDGES];
    atomicMax(&lp[node], i);
}

// ---------------------------------------------------------------------------
// Build message matrix [M_ROWS, K_MSG] (bf16, zero-padded) and h [M_ROWS, K_H].
// msg = [mem[a](500) | mem[b](500) | ef(1) | cos((t-lu[a])*tw + tb)(500) | 0pad]
// Reference's final enc uses (edge_times - last_update[node]) for BOTH
// directions (the un-shifted enc_s is dead code).
// ---------------------------------------------------------------------------
__global__ void build_msgs(const float* __restrict__ memory,
                           const float* __restrict__ last_update,
                           const float* __restrict__ edge_feat,
                           const float* __restrict__ edge_times,
                           const float* __restrict__ time_w,
                           const float* __restrict__ time_b,
                           const int* __restrict__ src,
                           const int* __restrict__ dst,
                           const int* __restrict__ edge_idxs,
                           bf16_t* __restrict__ msgs,
                           bf16_t* __restrict__ hbf)
{
    const int i = blockIdx.x;                       // 0..M_ROWS-1
    const int e = (i < B_EDGES) ? i : i - B_EDGES;
    const int a = (i < B_EDGES) ? src[e] : dst[e];  // node being updated
    const int b = (i < B_EDGES) ? dst[e] : src[e];
    const float t  = edge_times[e];
    const float dt = t - last_update[a];
    const float ef = edge_feat[edge_idxs[e]];
    const float* __restrict__ ma = memory + (size_t)a * DM;
    const float* __restrict__ mb = memory + (size_t)b * DM;
    bf16_t* __restrict__ mr = msgs + (size_t)i * K_MSG;
    bf16_t* __restrict__ hr = hbf + (size_t)i * K_H;

    for (int c = threadIdx.x; c < K_MSG; c += blockDim.x) {
        float v;
        if (c < DM)          v = ma[c];
        else if (c < 2 * DM) v = mb[c - DM];
        else if (c == 2 * DM) v = ef;
        else if (c < MSG) {
            int k = c - 2 * DM - 1;
            v = __cosf(dt * time_w[k] + time_b[k]);
        } else v = 0.0f;
        mr[c] = (bf16_t)v;
    }
    for (int c = threadIdx.x; c < K_H; c += blockDim.x)
        hr[c] = (bf16_t)((c < DM) ? ma[c] : 0.0f);
}

// ---------------------------------------------------------------------------
// GRU gates + scatter of last message per node.
// new_memory[node] = h_new exactly (delta-add collapses since h == memory[node]).
// ---------------------------------------------------------------------------
__global__ void gru_finalize(const float* __restrict__ gi, const float* __restrict__ gh,
                             const float* __restrict__ b_ih, const float* __restrict__ b_hh,
                             const float* __restrict__ memory,
                             const float* __restrict__ edge_times,
                             const int* __restrict__ src, const int* __restrict__ dst,
                             const int* __restrict__ lp,
                             float* __restrict__ out_mem, float* __restrict__ out_lu)
{
    const int i = blockIdx.x;
    const int e = (i < B_EDGES) ? i : i - B_EDGES;
    const int node = (i < B_EDGES) ? src[e] : dst[e];
    if (lp[node] != i) return;                      // only the last message writes

    const float* __restrict__ gir = gi + (size_t)i * NP;
    const float* __restrict__ ghr = gh + (size_t)i * NP;
    const float* __restrict__ h   = memory + (size_t)node * DM;
    float* __restrict__ om = out_mem + (size_t)node * DM;

    for (int c = threadIdx.x; c < DM; c += blockDim.x) {
        float xr = (gir[c] + b_ih[c]) + (ghr[c] + b_hh[c]);
        float xz = (gir[DM + c] + b_ih[DM + c]) + (ghr[DM + c] + b_hh[DM + c]);
        float r = 1.0f / (1.0f + __expf(-xr));
        float z = 1.0f / (1.0f + __expf(-xz));
        float n = tanhf((gir[2 * DM + c] + b_ih[2 * DM + c]) +
                        r * (ghr[2 * DM + c] + b_hh[2 * DM + c]));
        om[c] = (1.0f - z) * n + z * h[c];
    }
    if (threadIdx.x == 0) out_lu[node] = edge_times[e];
}

// ---------------------------------------------------------------------------
// Edge-score MLP (tiny FLOPs; VALU). Block b: edge b>>1, pos/neg = b&1.
// ---------------------------------------------------------------------------
__global__ __launch_bounds__(128) void mlp_edge(const float* __restrict__ memory,
                                                const float* __restrict__ ce,
                                                const float* __restrict__ W1,
                                                const float* __restrict__ b1,
                                                const float* __restrict__ W2,
                                                const float* __restrict__ b2,
                                                const float* __restrict__ W3,
                                                const float* __restrict__ b3,
                                                const int* __restrict__ src,
                                                const int* __restrict__ dst,
                                                const int* __restrict__ neg,
                                                float* __restrict__ out)
{
    __shared__ float xs[4 * DM];
    __shared__ float h1[80];
    __shared__ float h2[10];

    const int blk   = blockIdx.x;
    const int e     = blk >> 1;
    const int isNeg = blk & 1;
    const int s = src[e];
    const int o = isNeg ? neg[e] : dst[e];
    const float* __restrict__ ms = memory + (size_t)s * DM;
    const float* __restrict__ cs = ce + (size_t)s * DM;
    const float* __restrict__ mo = memory + (size_t)o * DM;
    const float* __restrict__ co = ce + (size_t)o * DM;

    for (int c = threadIdx.x; c < DM; c += blockDim.x) {
        xs[c]          = ms[c];
        xs[DM + c]     = cs[c];
        xs[2 * DM + c] = mo[c];
        xs[3 * DM + c] = co[c];
    }
    __syncthreads();
    if (threadIdx.x < 80) {
        float acc = b1[threadIdx.x];
        for (int k = 0; k < 4 * DM; ++k)
            acc = fmaf(xs[k], W1[k * 80 + threadIdx.x], acc);
        h1[threadIdx.x] = fmaxf(acc, 0.0f);
    }
    __syncthreads();
    if (threadIdx.x < 10) {
        float acc = b2[threadIdx.x];
        for (int k = 0; k < 80; ++k)
            acc = fmaf(h1[k], W2[k * 10 + threadIdx.x], acc);
        h2[threadIdx.x] = fmaxf(acc, 0.0f);
    }
    __syncthreads();
    if (threadIdx.x == 0) {
        float acc = b3[0];
        for (int k = 0; k < 10; ++k) acc = fmaf(h2[k], W3[k], acc);
        out[isNeg * B_EDGES + e] = acc;   // score_pos at [0..B), score_neg at [B..2B)
    }
}

// ---------------------------------------------------------------------------
extern "C" void kernel_launch(void* const* d_in, const int* in_sizes, int n_in,
                              void* d_out, int out_size, void* d_ws, size_t ws_size,
                              hipStream_t stream)
{
    const float* memory      = (const float*)d_in[0];
    const float* last_update = (const float*)d_in[1];
    const float* community   = (const float*)d_in[2];
    const float* edge_feat   = (const float*)d_in[3];
    const float* edge_times  = (const float*)d_in[4];
    const float* time_w      = (const float*)d_in[5];
    const float* time_b      = (const float*)d_in[6];
    const float* W_ih        = (const float*)d_in[7];
    const float* W_hh        = (const float*)d_in[8];
    const float* b_ih        = (const float*)d_in[9];
    const float* b_hh        = (const float*)d_in[10];
    const float* W1          = (const float*)d_in[11];
    const float* b1          = (const float*)d_in[12];
    const float* W2          = (const float*)d_in[13];
    const float* b2          = (const float*)d_in[14];
    const float* W3          = (const float*)d_in[15];
    const float* b3          = (const float*)d_in[16];
    const int*   src         = (const int*)d_in[17];
    const int*   dst         = (const int*)d_in[18];
    const int*   neg         = (const int*)d_in[19];
    const int*   edge_idxs   = (const int*)d_in[20];
    (void)in_sizes; (void)n_in; (void)out_size; (void)ws_size;

    // Output layout: score_pos[B] | score_neg[B] | new_memory[N*DM] | new_last_update[N]
    float* out     = (float*)d_out;
    float* out_mem = out + 2 * B_EDGES;
    float* out_lu  = out_mem + (size_t)N_NODES * DM;

    // Workspace carve-up (all offsets multiples of 256 B; total ~141 MB)
    char* ws = (char*)d_ws;
    size_t off = 0;
    bf16_t* msgs = (bf16_t*)(ws + off); off += (size_t)M_ROWS * K_MSG * sizeof(bf16_t);
    bf16_t* hbf  = (bf16_t*)(ws + off); off += (size_t)M_ROWS * K_H   * sizeof(bf16_t);
    bf16_t* wih  = (bf16_t*)(ws + off); off += (size_t)NP     * K_MSG * sizeof(bf16_t);
    bf16_t* whh  = (bf16_t*)(ws + off); off += (size_t)NP     * K_H   * sizeof(bf16_t);
    float*  gi   = (float*)(ws + off);  off += (size_t)M_ROWS * NP    * sizeof(float);
    float*  gh   = (float*)(ws + off);  off += (size_t)M_ROWS * NP    * sizeof(float);
    int*    lp   = (int*)(ws + off);    off += (size_t)N_NODES * sizeof(int);

    // 1) Bulk copy (HBM-bound floor ~34us @ 23.3 TB/s); scatter overrides later.
    hipMemcpyAsync(out_mem, memory, (size_t)N_NODES * DM * sizeof(float),
                   hipMemcpyDeviceToDevice, stream);
    hipMemcpyAsync(out_lu, last_update, (size_t)N_NODES * sizeof(float),
                   hipMemcpyDeviceToDevice, stream);

    // 2) 'last' aggregation
    init_lastpos<<<(N_NODES + 255) / 256, 256, 0, stream>>>(lp);
    scan_last<<<(M_ROWS + 255) / 256, 256, 0, stream>>>(src, dst, lp);

    // 3) bf16 weight conversion (L2-resident afterwards)
    conv_w<<<1024, 256, 0, stream>>>(W_ih, wih, 3 * DM, MSG, NP, K_MSG);
    conv_w<<<1024, 256, 0, stream>>>(W_hh, whh, 3 * DM, DM, NP, K_H);

    // 4) Gather + time-encode into bf16 message/h matrices
    build_msgs<<<M_ROWS, 256, 0, stream>>>(memory, last_update, edge_feat, edge_times,
                                           time_w, time_b, src, dst, edge_idxs, msgs, hbf);

    // 5) WMMA GEMMs: gi = msgs @ W_ih^T, gh = h @ W_hh^T  (~50 GFLOP total)
    //    64x64 block tile (4 waves x 16x64), software-pipelined K loop.
    dim3 grid_g(M_ROWS / 64, NP / 64);      // 128 x 24
    wmma_gemm_nt<K_MSG><<<grid_g, 128, 0, stream>>>(msgs, wih, gi);
    wmma_gemm_nt<K_H>  <<<grid_g, 128, 0, stream>>>(hbf, whh, gh);

    // 6) GRU gates + sparse scatter into copied memory
    gru_finalize<<<M_ROWS, 256, 0, stream>>>(gi, gh, b_ih, b_hh, memory, edge_times,
                                             src, dst, lp, out_mem, out_lu);

    // 7) Edge scores
    mlp_edge<<<2 * B_EDGES, 128, 0, stream>>>(memory, community, W1, b1, W2, b2,
                                              W3, b3, src, dst, neg, out);
}
